// CascadeAttention_4157528342786
// MI455X (gfx1250) — compile-verified, matrix operand order checked by reading, not run on hardware
//
#include <hip/hip_runtime.h>

// ---------------------------------------------------------------------------
// CascadeAttention for MI455X (gfx1250): bf16 WMMA GEMMs + flash attention.
// B=1, Q=2048, D=4096, H=32, KVH=8, HD=128, NS=4, NW=2048, L=4100.
// ---------------------------------------------------------------------------

#define QLEN  2048
#define DMODEL 4096
#define NHEAD 32
#define NKVH  8
#define HEADD 128
#define NSINK 4
#define NWIN  2048
#define LTOT  (NSINK + NWIN + QLEN)   // 4100
#define CSTART (NSINK + NWIN)         // 2052
#define KT    64                      // attention key tile

typedef __bf16 bf16_t;
typedef bf16_t v16bf __attribute__((ext_vector_type(16)));
typedef float  v8f   __attribute__((ext_vector_type(8)));

union Frag {
  v16bf v;
  unsigned int u[8];
};

__device__ __forceinline__ unsigned short f2bf(float f) {
  unsigned int u = __float_as_uint(f);
  u = (u + 0x7FFFu + ((u >> 16) & 1u)) >> 16;   // round-to-nearest-even
  return (unsigned short)u;
}

// packed f32x2 -> bf16x2 in one dword (hardware cvt when available)
__device__ __forceinline__ unsigned int f2bf2(float lo, float hi) {
#if __has_builtin(__builtin_amdgcn_cvt_pk_bf16_f32)
  typedef bf16_t v2bf __attribute__((ext_vector_type(2)));
  v2bf p = __builtin_amdgcn_cvt_pk_bf16_f32(lo, hi);
  return __builtin_bit_cast(unsigned int, p);
#else
  return (unsigned int)f2bf(lo) | ((unsigned int)f2bf(hi) << 16);
#endif
}

// K-index base for fragment dword j given lane half (ISA 16-bit A layout).
__device__ __forceinline__ int kk_of(int j, int half) {
  return (j < 4) ? ((half << 3) + (j << 1)) : (16 + (half << 3) + ((j - 4) << 1));
}

__device__ __forceinline__ v8f wmma_bf16(const Frag& a, const Frag& b, v8f c) {
  return __builtin_amdgcn_wmma_f32_16x16x32_bf16(false, a.v, false, b.v,
                                                 (short)0, c, false, false);
}

// ---------------------------------------------------------------------------
// GEMM: C[M,N] (f32) = A[M,Kd] x B[Kd,N].  A is f32 (converted) or bf16.
// Block: 256 threads = 8 waves, tile 128x128, K-step 32.
// ---------------------------------------------------------------------------
template <bool ABF16>
__global__ void __launch_bounds__(256)
gemm_wmma(const void* __restrict__ Aptr, const float* __restrict__ Bptr,
          float* __restrict__ Cptr, int M, int N, int Kd) {
  __shared__ unsigned short As[128][34];   // [row][k]
  __shared__ unsigned short Bs[128][34];   // [col][k]  (transposed W tile)

  const int tid  = threadIdx.x;
  const int wave = tid >> 5;
  const int lane = tid & 31;
  const int half = lane >> 4;
  const int r    = lane & 15;
  const int m0   = blockIdx.y * 128;
  const int n0   = blockIdx.x * 128;
  const int mrow = wave * 16;

  v8f acc[8];
  for (int o = 0; o < 8; ++o)
    for (int e = 0; e < 8; ++e) acc[o][e] = 0.f;

  for (int k0 = 0; k0 < Kd; k0 += 32) {
    __syncthreads();
    if (ABF16) {
      const unsigned short* A = (const unsigned short*)Aptr;
      for (int it = 0; it < 2; ++it) {
        int fid = tid + it * 256;               // 0..511
        int row = fid >> 2;
        int c8  = (fid & 3) << 3;
        uint4 dv = *(const uint4*)&A[(size_t)(m0 + row) * Kd + k0 + c8];
        const unsigned int* du = (const unsigned int*)&dv;
        for (int e = 0; e < 4; ++e)
          *(unsigned int*)&As[row][c8 + 2 * e] = du[e];
      }
    } else {
      const float* A = (const float*)Aptr;
      for (int it = 0; it < 4; ++it) {
        int fid = tid + it * 256;               // 0..1023
        int row = fid >> 3;
        int c4  = (fid & 7) << 2;
        float4 fv = *(const float4*)&A[(size_t)(m0 + row) * Kd + k0 + c4];
        *(unsigned int*)&As[row][c4 + 0] = f2bf2(fv.x, fv.y);
        *(unsigned int*)&As[row][c4 + 2] = f2bf2(fv.z, fv.w);
      }
    }
    for (int it = 0; it < 4; ++it) {
      int fid  = tid + it * 256;                // 0..1023
      int krow = fid >> 5;
      int c4   = (fid & 31) << 2;
      float4 fv = *(const float4*)&Bptr[(size_t)(k0 + krow) * N + n0 + c4];
      Bs[c4 + 0][krow] = f2bf(fv.x);
      Bs[c4 + 1][krow] = f2bf(fv.y);
      Bs[c4 + 2][krow] = f2bf(fv.z);
      Bs[c4 + 3][krow] = f2bf(fv.w);
    }
    __syncthreads();

    Frag a;
    for (int j = 0; j < 8; ++j)
      a.u[j] = *(const unsigned int*)&As[mrow + r][kk_of(j, half)];
    for (int o = 0; o < 8; ++o) {
      Frag b;
      for (int j = 0; j < 8; ++j)
        b.u[j] = *(const unsigned int*)&Bs[o * 16 + r][kk_of(j, half)];
      acc[o] = wmma_bf16(a, b, acc[o]);
    }
  }

  for (int o = 0; o < 8; ++o)
    for (int i = 0; i < 8; ++i)
      Cptr[(size_t)(m0 + mrow + i + 8 * half) * N + n0 + o * 16 + r] = acc[o][i];
}

// ---------------------------------------------------------------------------
// Bias (cache masks -> additive bias over full key length)
// ---------------------------------------------------------------------------
__global__ void bias_kernel(const float* __restrict__ sink_mask,
                            const float* __restrict__ key_mask,
                            float* __restrict__ bias) {
  int i = blockIdx.x * blockDim.x + threadIdx.x;
  if (i >= LTOT) return;
  float m = 0.f;
  if (i < NSINK) m = sink_mask[i];
  else if (i < CSTART) m = key_mask[i - NSINK];
  bias[i] = m * -3.4028234663852886e38f;
}

// ---------------------------------------------------------------------------
// RoPE kernels: f32 -> bf16 with rotation.  inv_freq = 1e4^(-d/64).
// ---------------------------------------------------------------------------
#define LOG1E4 9.210340371976184f

// src token-major [S, nh*HD]; dst head-major [nh, Ldst, HD]; pos = pbase + s
__global__ void rope_tok_kernel(const float* __restrict__ src,
                                unsigned short* __restrict__ dst,
                                int S, int nh, int Ldst, int seq_off,
                                const int* __restrict__ sink_pos,
                                const int* __restrict__ key_pos) {
  int tid = blockIdx.x * blockDim.x + threadIdx.x;
  if (tid >= S * nh * 64) return;
  int d = tid & 63;
  int h = (tid >> 6) % nh;
  int s = tid / (64 * nh);
  int a = sink_pos[NSINK - 1], b = key_pos[NWIN - 1];
  int pbase = ((a > b) ? a : b) + 1;
  float pos = (float)(pbase + s);
  float inv = __expf(-(float)d * (LOG1E4 / 64.f));
  float sn, cs;
  __sincosf(pos * inv, &sn, &cs);
  const float* row = src + (size_t)s * nh * HEADD + h * HEADD;
  float x1 = row[d], x2 = row[d + 64];
  unsigned short* orow = dst + ((size_t)h * Ldst + seq_off + s) * HEADD;
  orow[d]      = f2bf(x1 * cs - x2 * sn);
  orow[d + 64] = f2bf(x2 * cs + x1 * sn);
}

// src head-major [nh, S, HD]; positions from pos_arr[s]
__global__ void rope_head_kernel(const float* __restrict__ src,
                                 unsigned short* __restrict__ dst,
                                 int S, int nh, int Ldst, int seq_off,
                                 const int* __restrict__ pos_arr) {
  int tid = blockIdx.x * blockDim.x + threadIdx.x;
  if (tid >= S * nh * 64) return;
  int d = tid & 63;
  int h = (tid >> 6) % nh;
  int s = tid / (64 * nh);
  float pos = (float)pos_arr[s];
  float inv = __expf(-(float)d * (LOG1E4 / 64.f));
  float sn, cs;
  __sincosf(pos * inv, &sn, &cs);
  const float* row = src + ((size_t)h * S + s) * HEADD;
  float x1 = row[d], x2 = row[d + 64];
  unsigned short* orow = dst + ((size_t)h * Ldst + seq_off + s) * HEADD;
  orow[d]      = f2bf(x1 * cs - x2 * sn);
  orow[d + 64] = f2bf(x2 * cs + x1 * sn);
}

// V converts (no rotation)
__global__ void conv_tok_kernel(const float* __restrict__ src,
                                unsigned short* __restrict__ dst,
                                int S, int nh, int Ldst, int seq_off) {
  int tid = blockIdx.x * blockDim.x + threadIdx.x;
  if (tid >= S * nh * HEADD) return;
  int d = tid & 127;
  int h = (tid >> 7) % nh;
  int s = tid / (HEADD * nh);
  dst[((size_t)h * Ldst + seq_off + s) * HEADD + d] =
      f2bf(src[(size_t)s * nh * HEADD + h * HEADD + d]);
}

__global__ void conv_head_kernel(const float* __restrict__ src,
                                 unsigned short* __restrict__ dst,
                                 int S, int nh, int Ldst, int seq_off) {
  int tid = blockIdx.x * blockDim.x + threadIdx.x;
  if (tid >= S * nh * HEADD) return;
  int d = tid & 127;
  int h = (tid >> 7) % nh;
  int s = tid / (HEADD * nh);
  dst[((size_t)h * Ldst + seq_off + s) * HEADD + d] =
      f2bf(src[((size_t)h * S + s) * HEADD + d]);
}

// ---------------------------------------------------------------------------
// Flash attention: block = (q-tile of 128, head); 8 waves x 16 query rows.
// Online softmax over keys in 64-key steps; WMMA for QK^T and PV.
// Causal upper bound: keys beyond CSTART + qblock_end are never touched.
// ---------------------------------------------------------------------------
__global__ void __launch_bounds__(256)
attn_kernel(const unsigned short* __restrict__ qbf,
            const unsigned short* __restrict__ Kbf,
            const unsigned short* __restrict__ Vbf,
            const float* __restrict__ bias,
            unsigned short* __restrict__ obf) {
  __shared__ unsigned short Vs[128][KT + 2];      // [hd][key_local]
  __shared__ unsigned short Ps[8][16][KT + 2];    // per-wave P tile [row][key]

  const int tid  = threadIdx.x;
  const int wave = tid >> 5;
  const int lane = tid & 31;
  const int half = lane >> 4;
  const int r    = lane & 15;
  const int h    = blockIdx.y;
  const int kvh  = h >> 2;                        // G = H/KVH = 4
  const int q0   = blockIdx.x * 128 + wave * 16;
  const float SCALE = 0.08838834764831845f;       // 1/sqrt(128)

  // Q fragments for this wave's 16 rows (HD=128 -> 4 K-steps of 32)
  Frag qf[4];
  for (int kc = 0; kc < 4; ++kc) {
    const unsigned short* qp =
        &qbf[((size_t)h * QLEN + (q0 + r)) * HEADD + kc * 32];
    for (int j = 0; j < 8; ++j)
      qf[kc].u[j] = *(const unsigned int*)&qp[kk_of(j, half)];
  }

  float m_i[8], l_i[8];
  v8f O[8];
  for (int i = 0; i < 8; ++i) { m_i[i] = -1e30f; l_i[i] = 0.f; }
  for (int o = 0; o < 8; ++o)
    for (int e = 0; e < 8; ++e) O[o][e] = 0.f;

  // last key this block can ever see (causal): CSTART + qblock_end
  const int limit  = min(LTOT, CSTART + (int)blockIdx.x * 128 + 128);
  const int nsteps = (limit + KT - 1) / KT;

  for (int st = 0; st < nsteps; ++st) {
    const int kb = st * KT;
    __syncthreads();

    // stage V tile transposed: Vs[hd][key]  (64 keys x 128 hd)
    for (int it = 0; it < 4; ++it) {
      int idx = tid + it * 256;                   // 0..1023
      int kl  = idx >> 4;                         // 0..63
      int h8  = (idx & 15) << 3;
      int key = kb + kl;
      uint4 dv = make_uint4(0u, 0u, 0u, 0u);
      if (key < LTOT)
        dv = *(const uint4*)&Vbf[((size_t)kvh * LTOT + key) * HEADD + h8];
      const unsigned short* ds = (const unsigned short*)&dv;
      for (int e = 0; e < 8; ++e) Vs[h8 + e][kl] = ds[e];
    }

    // scores: S = q (16x128) @ K^T (128x64) -> 4 column tiles of 16
    v8f sc[4];
    for (int nt = 0; nt < 4; ++nt) {
      for (int e = 0; e < 8; ++e) sc[nt][e] = 0.f;
      int key = kb + nt * 16 + r;
      bool ok = key < LTOT;
      const unsigned short* kp =
          &Kbf[((size_t)kvh * LTOT + (ok ? key : 0)) * HEADD];
      for (int kc = 0; kc < 4; ++kc) {
        Frag b;
        for (int j = 0; j < 8; ++j)
          b.u[j] = ok ? *(const unsigned int*)&kp[kc * 32 + kk_of(j, half)] : 0u;
        sc[nt] = wmma_bf16(qf[kc], b, sc[nt]);
      }
    }

    // masks + bias (row = i + 8*half, col = nt*16 + r); interior steps skip
    // the per-element mask test entirely.
    const bool edge = (kb + KT > LTOT) || (kb + KT > CSTART + q0 + 1);
    float sv[4][8];
    for (int nt = 0; nt < 4; ++nt) {
      int key = kb + nt * 16 + r;
      float bv = (key < LTOT) ? bias[key] : 0.f;
      for (int i = 0; i < 8; ++i) {
        float v = sc[nt][i] * SCALE + bv;
        if (edge) {
          int qi = q0 + i + 8 * half;
          if (key >= LTOT || (key >= CSTART && (key - CSTART) > qi))
            v = -3.0e38f;
        }
        sv[nt][i] = v;
      }
    }

    // online softmax per row; write P tile (bf16) for A-fragment transpose
    for (int i = 0; i < 8; ++i) {
      float mx = fmaxf(fmaxf(sv[0][i], sv[1][i]), fmaxf(sv[2][i], sv[3][i]));
      for (int d = 1; d < 16; d <<= 1) mx = fmaxf(mx, __shfl_xor(mx, d, 32));
      float nm = fmaxf(m_i[i], mx);
      float fs = __expf(m_i[i] - nm);
      float rs = 0.f;
      for (int nt = 0; nt < 4; ++nt) {
        float p = __expf(sv[nt][i] - nm);
        rs += p;
        Ps[wave][i + 8 * half][nt * 16 + r] = f2bf(p);
      }
      for (int d = 1; d < 16; d <<= 1) rs += __shfl_xor(rs, d, 32);
      l_i[i] = l_i[i] * fs + rs;
      m_i[i] = nm;
      for (int o = 0; o < 8; ++o) O[o][i] *= fs;
    }

    __syncthreads();

    // PV: O += P (16x64) @ V (64x128): two 32-key halves
    Frag pa0, pa1;
    for (int j = 0; j < 8; ++j) {
      int kk = kk_of(j, half);
      pa0.u[j] = *(const unsigned int*)&Ps[wave][r][kk];
      pa1.u[j] = *(const unsigned int*)&Ps[wave][r][32 + kk];
    }
    for (int o = 0; o < 8; ++o) {
      Frag vb0, vb1;
      for (int j = 0; j < 8; ++j) {
        int kk = kk_of(j, half);
        vb0.u[j] = *(const unsigned int*)&Vs[o * 16 + r][kk];
        vb1.u[j] = *(const unsigned int*)&Vs[o * 16 + r][32 + kk];
      }
      O[o] = wmma_bf16(pa0, vb0, O[o]);
      O[o] = wmma_bf16(pa1, vb1, O[o]);
    }
  }

  // normalize and store bf16 attention output [Q, H*HD]
  for (int o = 0; o < 8; ++o)
    for (int i = 0; i < 8; ++i) {
      int qi  = q0 + i + 8 * half;
      int col = o * 16 + r;
      obf[(size_t)qi * (NHEAD * HEADD) + h * HEADD + col] =
          f2bf(O[o][i] / l_i[i]);
    }
}

// ---------------------------------------------------------------------------
// Host orchestration
// ---------------------------------------------------------------------------
extern "C" void kernel_launch(void* const* d_in, const int* in_sizes, int n_in,
                              void* d_out, int out_size, void* d_ws, size_t ws_size,
                              hipStream_t stream) {
  (void)in_sizes; (void)n_in; (void)out_size; (void)ws_size;

  const float* hidden    = (const float*)d_in[0];
  const float* sink_k    = (const float*)d_in[1];
  const float* sink_v    = (const float*)d_in[2];
  const float* win_k     = (const float*)d_in[3];
  const float* win_v     = (const float*)d_in[4];
  const int*   sink_pos  = (const int*)d_in[5];
  const int*   key_pos   = (const int*)d_in[6];
  const float* sink_mask = (const float*)d_in[7];
  const float* key_mask  = (const float*)d_in[8];
  const float* Wq        = (const float*)d_in[9];
  const float* Wk        = (const float*)d_in[10];
  const float* Wv        = (const float*)d_in[11];
  const float* Wo        = (const float*)d_in[12];

  char* ws = (char*)d_ws;
  float*          q_f32  = (float*)(ws);                         // 32 MB
  float*          k_f32  = (float*)(ws + 33554432);              //  8 MB
  float*          v_f32  = (float*)(ws + 41943040);              //  8 MB
  unsigned short* q_bf   = (unsigned short*)(ws + 50331648);     // 16 MB
  unsigned short* K_bf   = (unsigned short*)(ws + 67108864);     // 8396800 B
  unsigned short* V_bf   = (unsigned short*)(ws + 75505664);     // 8396800 B
  float*          biasb  = (float*)(ws + 83902464);              // 16400 B
  unsigned short* attn_bf = (unsigned short*)q_f32;              // alias (q_f32 dead)

  // QKV projections (fp32 -> bf16 WMMA, fp32 accumulate)
  gemm_wmma<false><<<dim3(32, 16), 256, 0, stream>>>(hidden, Wq, q_f32, QLEN, 4096, DMODEL);
  gemm_wmma<false><<<dim3(8, 16),  256, 0, stream>>>(hidden, Wk, k_f32, QLEN, 1024, DMODEL);
  gemm_wmma<false><<<dim3(8, 16),  256, 0, stream>>>(hidden, Wv, v_f32, QLEN, 1024, DMODEL);

  bias_kernel<<<(LTOT + 255) / 256, 256, 0, stream>>>(sink_mask, key_mask, biasb);

  // RoPE + layout conversion to head-major bf16
  rope_tok_kernel<<<(QLEN * NHEAD * 64) / 256, 256, 0, stream>>>(
      q_f32, q_bf, QLEN, NHEAD, QLEN, 0, sink_pos, key_pos);
  rope_tok_kernel<<<(QLEN * NKVH * 64) / 256, 256, 0, stream>>>(
      k_f32, K_bf, QLEN, NKVH, LTOT, CSTART, sink_pos, key_pos);
  rope_head_kernel<<<(NSINK * NKVH * 64 + 255) / 256, 256, 0, stream>>>(
      sink_k, K_bf, NSINK, NKVH, LTOT, 0, sink_pos);
  rope_head_kernel<<<(NWIN * NKVH * 64) / 256, 256, 0, stream>>>(
      win_k, K_bf, NWIN, NKVH, LTOT, NSINK, key_pos);
  conv_tok_kernel<<<(QLEN * NKVH * HEADD) / 256, 256, 0, stream>>>(
      v_f32, V_bf, QLEN, NKVH, LTOT, CSTART);
  conv_head_kernel<<<(NSINK * NKVH * HEADD + 255) / 256, 256, 0, stream>>>(
      sink_v, V_bf, NSINK, NKVH, LTOT, 0);
  conv_head_kernel<<<(NWIN * NKVH * HEADD) / 256, 256, 0, stream>>>(
      win_v, V_bf, NWIN, NKVH, LTOT, NSINK);

  // Flash attention with GQA (kv head = h/4), causal on current block
  attn_kernel<<<dim3(QLEN / 128, NHEAD), 256, 0, stream>>>(q_bf, K_bf, V_bf, biasb, attn_bf);

  // Output projection -> fp32 d_out
  gemm_wmma<true><<<dim3(32, 16), 256, 0, stream>>>(attn_bf, Wo, (float*)d_out,
                                                    QLEN, DMODEL, NHEAD * HEADD);
}
